// Verification_BiLSTM_CRF_allennlp_46454366273893
// MI455X (gfx1250) — compile-verified, matrix operand order checked by reading, not run on hardware
//
#include <hip/hip_runtime.h>

typedef __bf16 bhalf;
typedef __attribute__((ext_vector_type(16))) __bf16 v16bf;
typedef __attribute__((ext_vector_type(8)))  float  v8f;

#define ACT_NONE 0
#define ACT_TANH 1
#define ACT_RELU 2

__device__ __forceinline__ bhalf f2bf(float f) {
  union { float f; unsigned u; } v; v.f = f;
  unsigned r = (v.u + 0x7FFFu + ((v.u >> 16) & 1u)) >> 16;
  union { unsigned short s; bhalf b; } o; o.s = (unsigned short)r;
  return o.b;
}

__device__ __forceinline__ float sigm(float x) { return 1.0f / (1.0f + __expf(-x)); }

// ---------------------------------------------------------------------------
// WMMA fragment layout (ISA 7.12.2, 16-bit operands, wave32):
//   fragment element (lane l, elem e) <-> K = kt*32 + ((l>>4)<<3) + (e<8? e : e+8)
//   A: row = l&15 ;  B: col = l&15
// Packed storage (both A-in-LDS and B-in-global): tile kt -> 512 bf16 at
//   [kt*512 + l*16 + e]  (B additionally indexed by nt: tile (kt,nt)).
// ---------------------------------------------------------------------------

// Cooperative pack of A (Mrows x K f32, row-major lda) into bf16 fragments.
__device__ __forceinline__ void pack_A(const float* A, int lda, int Mrows, int ktiles,
                                       bhalf* dst) {
  const int total = ktiles << 9;
  for (int i = threadIdx.x; i < total; i += 256) {
    int kt = i >> 9, rem = i & 511, l = rem >> 4, e = rem & 15;
    int row = l & 15;
    int k = kt * 32 + ((l >> 4) << 3) + (e < 8 ? e : e + 8);
    float v = (row < Mrows) ? A[(size_t)row * lda + k] : 0.0f;
    dst[i] = f2bf(v);
  }
}

// Scatter one (row b, col k) value into packed-A fragments (b must be < 8).
__device__ __forceinline__ void pack_elem(bhalf* dst, int b, int k, float v) {
  int kt = k >> 5, km = k & 31;
  int l, e;
  if (km < 8)       { l = b;      e = km; }
  else if (km < 16) { l = b + 16; e = km - 8; }
  else if (km < 24) { l = b;      e = km - 8; }
  else              { l = b + 16; e = km - 16; }
  dst[(kt << 9) + (l << 4) + e] = f2bf(v);
}

// acc += Apacked * Wpacked(:, nt)   -- pure load + wmma inner loop.
__device__ __forceinline__ void wmma_acc_p(const bhalf* Ap, const bhalf* Wt,
                                           int ktiles, int ntiles, int nt, v8f& acc) {
  const int lane = threadIdx.x & 31;
  for (int kt = 0; kt < ktiles; ++kt) {
    v16bf a = *(const v16bf*)(Ap + (kt << 9) + (lane << 4));
    v16bf b = *(const v16bf*)(Wt + (((size_t)kt * ntiles + nt) << 9) + (lane << 4));
    acc = __builtin_amdgcn_wmma_f32_16x16x32_bf16(false, a, false, b, (short)0, acc,
                                                  false, false);
  }
}

__device__ __forceinline__ void wmma_store(v8f acc, float* D, int ldd, int Mrows, int nbase,
                                           const float* bias, int act) {
  const int lane = threadIdx.x & 31;
  const int col = lane & 15;
  const int mbase = (lane >> 4) << 3;
  float bv = bias ? bias[nbase + col] : 0.0f;
#pragma unroll
  for (int r = 0; r < 8; ++r) {
    int m = mbase + r;
    if (m < Mrows) {
      float v = acc[r] + bv;
      if (act == ACT_TANH) v = tanhf(v);
      else if (act == ACT_RELU) v = fmaxf(v, 0.0f);
      D[(size_t)m * ldd + nbase + col] = v;
    }
  }
}

__device__ __forceinline__ v8f wmma_zero() {
  v8f acc;
#pragma unroll
  for (int r = 0; r < 8; ++r) acc[r] = 0.0f;
  return acc;
}

// ---------------- weight packing: f32 (K x N row-major) -> bf16 WMMA B tiles ---------------
__global__ __launch_bounds__(256) void k_pack(const float* __restrict__ src,
                                              bhalf* __restrict__ dst, int K, int N) {
  int idx = blockIdx.x * 256 + threadIdx.x;
  int total = K * N;
  if (idx >= total) return;
  int ntiles = N >> 4;
  int tile = idx >> 9, rem = idx & 511, l = rem >> 4, e = rem & 15;
  int kt = tile / ntiles, nt = tile % ntiles;
  int Krow = kt * 32 + ((l >> 4) << 3) + (e < 8 ? e : e + 8);
  int Ncol = nt * 16 + (l & 15);
  dst[idx] = f2bf(src[(size_t)Krow * N + Ncol]);
}

// ---------------- embedding gathers: ea (128,8,512) and decoder token embs (2,32,8,512) ----
__global__ __launch_bounds__(256) void k_embed(const int* __restrict__ article,
                                               const int* __restrict__ summary,
                                               const float* __restrict__ emb,
                                               float* __restrict__ ea,
                                               float* __restrict__ es) {
  size_t idx = (size_t)blockIdx.x * 256 + threadIdx.x;
  if (idx < (size_t)128 * 8 * 512) {
    size_t row = idx >> 9; int k = (int)(idx & 511);
    int tok = article[row];
    ea[idx] = emb[(size_t)tok * 512 + k];
  } else if (idx < (size_t)128 * 8 * 512 + (size_t)2 * 32 * 8 * 512) {
    size_t r = idx - (size_t)128 * 8 * 512;
    size_t row = r >> 9; int k = (int)(r & 511);
    int dir = (int)(row >> 8);
    int sb = (int)(row & 255); int s = sb >> 3, b = sb & 7;
    int si = (dir == 0) ? ((s == 0) ? 0 : s - 1) : ((s == 0) ? 31 : 32 - s);
    int tok = summary[si * 8 + b];
    es[r] = emb[(size_t)tok * 512 + k];
  }
}

// ---------------- bidirectional LSTM encoder: grid=2 (dir), 256 threads = 8 waves ----------
__global__ __launch_bounds__(256) void k_encoder(const float* __restrict__ ea,
                                                 const bhalf* pWih_f, const bhalf* pWhh_f, const float* b_f,
                                                 const bhalf* pWih_b, const bhalf* pWhh_b, const float* b_b,
                                                 float* __restrict__ enc_out, float* __restrict__ hfin) {
  const int dir = blockIdx.x;
  const bhalf* pWih = dir ? pWih_b : pWih_f;
  const bhalf* pWhh = dir ? pWhh_b : pWhh_f;
  const float* bias = dir ? b_b : b_f;
  __shared__ float hs[8][256];
  __shared__ float cs[8][256];
  __shared__ float gs[8][1024];
  __shared__ bhalf apx[16 * 512];   // packed x fragments  (K=512 -> 16 ktiles)
  __shared__ bhalf aph[8 * 512];    // packed h fragments  (K=256 -> 8 ktiles)
  const int tid = threadIdx.x, wave = tid >> 5;
  for (int i = tid; i < 8 * 256; i += 256) { (&hs[0][0])[i] = 0.0f; (&cs[0][0])[i] = 0.0f; }
  __syncthreads();
  for (int t = 0; t < 128; ++t) {
    int li = dir ? (127 - t) : t;
    // pack this step's input + hidden state into WMMA fragments (once per step)
    pack_A(ea + (size_t)li * 8 * 512, 512, 8, 16, apx);
    pack_A(&hs[0][0], 256, 8, 8, aph);
    if (t + 1 < 128) {
      int ln = dir ? (126 - t) : (t + 1);
      __builtin_prefetch(ea + (size_t)ln * 8 * 512 + tid * 16, 0, 0);
    }
    __syncthreads();
    for (int nt = wave; nt < 64; nt += 8) {                 // N = 4*HE = 1024
      v8f acc = wmma_zero();
      wmma_acc_p(apx, pWih, 16, 64, nt, acc);
      wmma_acc_p(aph, pWhh, 8, 64, nt, acc);
      wmma_store(acc, &gs[0][0], 1024, 8, nt * 16, bias, ACT_NONE);
    }
    __syncthreads();
    for (int i = tid; i < 8 * 256; i += 256) {
      int b = i >> 8, j = i & 255;
      float ig = gs[b][j], fg = gs[b][256 + j], gg = gs[b][512 + j], og = gs[b][768 + j];
      float c2 = sigm(fg) * cs[b][j] + sigm(ig) * tanhf(gg);
      float h2 = sigm(og) * tanhf(c2);
      cs[b][j] = c2; hs[b][j] = h2;
      enc_out[((size_t)li * 8 + b) * 512 + dir * 256 + j] = h2;
    }
    __syncthreads();
  }
  for (int i = tid; i < 8 * 256; i += 256) hfin[dir * 8 * 256 + i] = (&hs[0][0])[i];
}

// ---------------- h0/c0 = tanh(hcat @ fc_{h,c}_W + b) -------------------------------------
__global__ __launch_bounds__(256) void k_h0c0(const float* __restrict__ hfin,
                                              const bhalf* p_fc_h, const float* fc_h_b,
                                              const bhalf* p_fc_c, const float* fc_c_b,
                                              float* __restrict__ h0, float* __restrict__ c0) {
  __shared__ float hc[8][512];
  __shared__ bhalf ap[16 * 512];
  const int tid = threadIdx.x, wave = tid >> 5;
  for (int i = tid; i < 8 * 512; i += 256) {
    int b = i >> 9, j = i & 511;
    hc[b][j] = (j < 256) ? hfin[b * 256 + j] : hfin[8 * 256 + b * 256 + (j - 256)];
  }
  __syncthreads();
  pack_A(&hc[0][0], 512, 8, 16, ap);
  __syncthreads();
  for (int job = wave; job < 64; job += 8) {
    int which = job >> 5, nt = job & 31;
    const bhalf* W = which ? p_fc_c : p_fc_h;
    const float* bias = which ? fc_c_b : fc_h_b;
    float* D = which ? c0 : h0;
    v8f acc = wmma_zero();
    wmma_acc_p(ap, W, 16, 32, nt, acc);
    wmma_store(acc, D, 512, 8, nt * 16, bias, ACT_TANH);
  }
}

// ---------------- hoisted attention precompute: EP[dir] = enc_out(1024x512) @ W_bot --------
// grid = 128 blocks: blockIdx.x = dir*64 + mt ; each block packs one 16-row A tile.
__global__ __launch_bounds__(256) void k_attnpre(const float* __restrict__ enc_out,
                                                 const bhalf* p_bot_f, const bhalf* p_bot_b,
                                                 float* __restrict__ EP) {
  __shared__ bhalf ap[16 * 512];
  const int tid = threadIdx.x, wave = tid >> 5;
  const int dir = blockIdx.x >> 6, mt = blockIdx.x & 63;
  const bhalf* W = dir ? p_bot_b : p_bot_f;
  float* D = EP + (size_t)dir * 1024 * 512 + (size_t)mt * 16 * 512;
  pack_A(enc_out + (size_t)mt * 16 * 512, 512, 16, 16, ap);
  __syncthreads();
  for (int nt = wave; nt < 32; nt += 8) {
    v8f acc = wmma_zero();
    wmma_acc_p(ap, W, 16, 32, nt, acc);
    wmma_store(acc, D, 512, 16, nt * 16, nullptr, ACT_NONE);
  }
}

// ---------------- attention decoders: grid=2 (dir), per-step hoisted attention -------------
__global__ __launch_bounds__(256) void k_decoder(const float* __restrict__ es,
                                                 const float* __restrict__ h0,
                                                 const float* __restrict__ c0,
                                                 const float* __restrict__ EP,
                                                 const float* __restrict__ enc_out,
                                                 const bhalf* p_top_f, const bhalf* p_top_b,
                                                 const float* v_f, const float* v_b,
                                                 const bhalf* pWih_f, const bhalf* pWih_b,
                                                 const bhalf* pWhh_f, const bhalf* pWhh_b,
                                                 const float* db_f, const float* db_b,
                                                 float* __restrict__ outs) {
  const int dir = blockIdx.x, tid = threadIdx.x, wave = tid >> 5;
  const bhalf* pTop = dir ? p_top_b : p_top_f;
  const float* av   = dir ? v_b : v_f;
  const bhalf* pWih = dir ? pWih_b : pWih_f;
  const bhalf* pWhh = dir ? pWhh_b : pWhh_f;
  const float* bias = dir ? db_b : db_f;
  const float* EPd  = EP + (size_t)dir * 1024 * 512;
  const float* esd  = es + (size_t)dir * 32 * 8 * 512;
  float* outd = outs + (size_t)dir * 32 * 8 * 512;

  __shared__ float hsd[8][512];
  __shared__ float csd[8][512];
  __shared__ bhalf aph[16 * 512];   // packed h fragments  (K=512)
  __shared__ bhalf apx[32 * 512];   // packed x fragments  (K=1024), built in-place
  __shared__ union UU {
    struct { float hp[8][512]; float sc[128][8]; } a;  // attention scratch
    float gb[8][2048];                                 // gate buffer (overlaid)
  } u;

  for (int i = tid; i < 8 * 512; i += 256) { (&hsd[0][0])[i] = h0[i]; (&csd[0][0])[i] = c0[i]; }
  for (int i = tid; i < 32 * 512; i += 256) apx[i] = f2bf(0.0f);   // dead lanes stay zero
  __syncthreads();

  for (int s = 0; s < 32; ++s) {
    // pack current hidden state once; used by both the attention and gate GEMMs
    pack_A(&hsd[0][0], 512, 8, 16, aph);
    __syncthreads();
    // hp = h @ W_top   (8x512 @ 512x512)
    for (int nt = wave; nt < 32; nt += 8) {
      v8f acc = wmma_zero();
      wmma_acc_p(aph, pTop, 16, 32, nt, acc);
      wmma_store(acc, &u.a.hp[0][0], 512, 8, nt * 16, nullptr, ACT_NONE);
    }
    __syncthreads();
    // score[l,b] = sum_j tanh(hp[b,j] + EP[l,b,j]) * v[j]
    for (int i = tid; i < 128 * 8; i += 256) {
      int l = i >> 3, b = i & 7;
      const float* ep = EPd + ((size_t)l * 8 + b) * 512;
      float ssum = 0.0f;
      for (int j = 0; j < 512; ++j) ssum += tanhf(u.a.hp[b][j] + ep[j]) * av[j];
      u.a.sc[l][b] = ssum;
    }
    __syncthreads();
    // softmax over l (per batch column)
    if (tid < 8) {
      int b = tid;
      float m = -1e30f;
      for (int l = 0; l < 128; ++l) m = fmaxf(m, u.a.sc[l][b]);
      float ssum = 0.0f;
      for (int l = 0; l < 128; ++l) { float e = __expf(u.a.sc[l][b] - m); u.a.sc[l][b] = e; ssum += e; }
      float inv = 1.0f / ssum;
      for (int l = 0; l < 128; ++l) u.a.sc[l][b] *= inv;
    }
    __syncthreads();
    // x = [token_emb, context] scattered straight into packed bf16 fragments
    for (int i = tid; i < 8 * 512; i += 256) {
      int b = i >> 9, j = i & 511;
      pack_elem(apx, b, j, esd[((size_t)s * 8 + b) * 512 + j]);
      float w = 0.0f;
      for (int l = 0; l < 128; ++l) w += u.a.sc[l][b] * enc_out[((size_t)l * 8 + b) * 512 + j];
      pack_elem(apx, b, 512 + j, w);
    }
    __syncthreads();
    // gates = x @ Wih + h @ Whh + b  (8x2048, K=1024 + K=512)
    for (int nt = wave; nt < 128; nt += 8) {
      v8f acc = wmma_zero();
      wmma_acc_p(apx, pWih, 32, 128, nt, acc);
      wmma_acc_p(aph, pWhh, 16, 128, nt, acc);
      wmma_store(acc, &u.gb[0][0], 2048, 8, nt * 16, bias, ACT_NONE);
    }
    __syncthreads();
    for (int i = tid; i < 8 * 512; i += 256) {
      int b = i >> 9, j = i & 511;
      float ig = u.gb[b][j], fg = u.gb[b][512 + j], gg = u.gb[b][1024 + j], og = u.gb[b][1536 + j];
      float c2 = sigm(fg) * csd[b][j] + sigm(ig) * tanhf(gg);
      float h2 = sigm(og) * tanhf(c2);
      csd[b][j] = c2; hsd[b][j] = h2;
      outd[((size_t)s * 8 + b) * 512 + j] = h2;
    }
    __syncthreads();
  }
}

// ---------------- hid = relu(combined @ fc2_W + b)  (256x1024 @ 1024x1024) -----------------
__global__ __launch_bounds__(256) void k_fc2(const float* __restrict__ outs,
                                             const bhalf* p_fc2, const float* fc2_b,
                                             float* __restrict__ hid) {
  __shared__ bhalf ap[32 * 512];    // packed combined tile, K=1024
  const int tid = threadIdx.x, wave = tid >> 5;
  const int mt = blockIdx.x;                       // 16 blocks, 16 rows each
  for (int i = tid; i < 32 * 512; i += 256) {
    int kt = i >> 9, rem = i & 511, l = rem >> 4, e = rem & 15;
    int r = l & 15;
    int k = kt * 32 + ((l >> 4) << 3) + (e < 8 ? e : e + 8);
    int m = mt * 16 + r, s = m >> 3, b = m & 7;
    float v = (k < 512)
                  ? outs[((size_t)s * 8 + b) * 512 + k]
                  : outs[(size_t)32 * 8 * 512 + ((size_t)(31 - s) * 8 + b) * 512 + (k - 512)];
    ap[i] = f2bf(v);
  }
  __syncthreads();
  for (int nt = wave; nt < 64; nt += 8) {
    v8f acc = wmma_zero();
    wmma_acc_p(ap, p_fc2, 32, 64, nt, acc);
    wmma_store(acc, hid + (size_t)mt * 16 * 1024, 1024, 16, nt * 16, fc2_b, ACT_RELU);
  }
}

// ---------------- feats = [combined, q] @ tag_W + tag_b, stored (B,S,T) --------------------
__global__ __launch_bounds__(256) void k_feats(const float* __restrict__ outs,
                                               const float* __restrict__ q,
                                               const float* __restrict__ tag_W,
                                               const float* __restrict__ tag_b,
                                               float* __restrict__ feats_out) {
  int idx = blockIdx.x * 256 + threadIdx.x;        // 2048 = 256 rows * 8 tags
  if (idx >= 2048) return;
  int row = idx >> 3, t = idx & 7, s = row >> 3, b = row & 7;
  const float* cf = outs + ((size_t)s * 8 + b) * 512;
  const float* cb = outs + (size_t)32 * 8 * 512 + ((size_t)(31 - s) * 8 + b) * 512;
  float acc = tag_b[t];
  for (int k = 0; k < 512; ++k) acc += cf[k] * tag_W[k * 8 + t];
  for (int k = 0; k < 512; ++k) acc += cb[k] * tag_W[(512 + k) * 8 + t];
  acc += q[s * 8 + b] * tag_W[1024 * 8 + t];
  feats_out[((size_t)b * 32 + s) * 8 + t] = acc;
}

// ---------------- output_binary[b] = mean_s(hid @ bin_W) + bin_b ---------------------------
__global__ __launch_bounds__(256) void k_bin(const float* __restrict__ hid,
                                             const float* __restrict__ bin_W,
                                             const float* __restrict__ bin_b,
                                             float* __restrict__ ob) {
  __shared__ float rd[256];
  const int tid = threadIdx.x;
  float acc = 0.0f;
  const float* hr = hid + (size_t)tid * 1024;
  for (int k = 0; k < 1024; ++k) acc += hr[k] * bin_W[k];
  rd[tid] = acc;
  __syncthreads();
  if (tid < 8) {
    float m = 0.0f;
    for (int s = 0; s < 32; ++s) m += rd[s * 8 + tid];
    ob[tid] = m / 32.0f + bin_b[0];
  }
}

// ---------------- constrained Viterbi, B=8, T=8, S=32 --------------------------------------
__global__ __launch_bounds__(32) void k_viterbi(const float* __restrict__ feats,   // (B,S,T)
                                                const int* __restrict__ summary,   // (S,B)
                                                const float* __restrict__ trans,
                                                const float* __restrict__ startv,
                                                const float* __restrict__ endv,
                                                float* __restrict__ tags_out,
                                                float* __restrict__ scores_out) {
  __shared__ int bps[8][31][8];
  const int b = threadIdx.x;
  if (b >= 8) return;
  const unsigned char allowed[8] = {0x01, 0x8E, 0x9E, 0xAE, 0x9E, 0xAE, 0x0E, 0x01};
  float tc[8][8];
  for (int i = 0; i < 8; ++i)
    for (int j = 0; j < 8; ++j)
      tc[i][j] = ((allowed[i] >> j) & 1) ? trans[i * 8 + j] : -10000.0f;
  const float* fb = feats + (size_t)b * 32 * 8;
  float score[8];
  for (int t = 0; t < 8; ++t) score[t] = startv[t] + fb[t];
  for (int s = 1; s < 32; ++s) {
    bool m = summary[(size_t)s * 8 + b] != 0;
    float best[8]; int bp[8];
    for (int j = 0; j < 8; ++j) {
      float bv = -1e30f; int bi = 0;
      for (int i = 0; i < 8; ++i) { float c = score[i] + tc[i][j]; if (c > bv) { bv = c; bi = i; } }
      best[j] = bv; bp[j] = bi;
    }
    for (int j = 0; j < 8; ++j) {
      bps[b][s - 1][j] = m ? bp[j] : j;
      if (m) score[j] = best[j] + fb[s * 8 + j];
    }
  }
  float fs = -1e30f; int lt = 0;
  for (int t = 0; t < 8; ++t) { float v = score[t] + endv[t]; if (v > fs) { fs = v; lt = t; } }
  scores_out[b] = fs;
  int path[32]; int tag = lt; path[31] = tag;
  for (int s = 30; s >= 0; --s) { tag = bps[b][s][tag]; path[s] = tag; }
  for (int s = 0; s < 32; ++s) tags_out[b * 32 + s] = (float)path[s];
}

extern "C" void kernel_launch(void* const* d_in, const int* in_sizes, int n_in,
                              void* d_out, int out_size, void* d_ws, size_t ws_size,
                              hipStream_t stream) {
  (void)in_sizes; (void)n_in; (void)out_size; (void)ws_size;
  const int*   article   = (const int*)d_in[0];
  const int*   summary   = (const int*)d_in[2];
  const float* qlabel    = (const float*)d_in[4];
  const float* emb       = (const float*)d_in[5];
  const float* enc_Wih_f = (const float*)d_in[6];
  const float* enc_Whh_f = (const float*)d_in[7];
  const float* enc_b_f   = (const float*)d_in[8];
  const float* enc_Wih_b = (const float*)d_in[9];
  const float* enc_Whh_b = (const float*)d_in[10];
  const float* enc_b_b   = (const float*)d_in[11];
  const float* fc_h_W    = (const float*)d_in[12];
  const float* fc_h_b    = (const float*)d_in[13];
  const float* fc_c_W    = (const float*)d_in[14];
  const float* fc_c_b    = (const float*)d_in[15];
  const float* attn_W_f  = (const float*)d_in[16];
  const float* attn_v_f  = (const float*)d_in[17];
  const float* dec_Wih_f = (const float*)d_in[18];
  const float* dec_Whh_f = (const float*)d_in[19];
  const float* dec_b_f   = (const float*)d_in[20];
  const float* attn_W_b  = (const float*)d_in[21];
  const float* attn_v_b  = (const float*)d_in[22];
  const float* dec_Wih_b = (const float*)d_in[23];
  const float* dec_Whh_b = (const float*)d_in[24];
  const float* dec_b_b   = (const float*)d_in[25];
  const float* tag_W     = (const float*)d_in[26];
  const float* tag_b     = (const float*)d_in[27];
  const float* fc2_W     = (const float*)d_in[28];
  const float* fc2_b     = (const float*)d_in[29];
  const float* bin_W     = (const float*)d_in[30];
  const float* bin_b     = (const float*)d_in[31];
  const float* crf_trans = (const float*)d_in[32];
  const float* crf_start = (const float*)d_in[33];
  const float* crf_end   = (const float*)d_in[34];

  char* ws = (char*)d_ws;
  size_t off = 0;
  auto alloc = [&](size_t bytes) -> char* {
    off = (off + 255) & ~(size_t)255;
    char* p = ws + off;
    off += bytes;
    return p;
  };

  // bf16 packed weights
  bhalf* p_enc_Wih_f = (bhalf*)alloc((size_t)512 * 1024 * 2);
  bhalf* p_enc_Whh_f = (bhalf*)alloc((size_t)256 * 1024 * 2);
  bhalf* p_enc_Wih_b = (bhalf*)alloc((size_t)512 * 1024 * 2);
  bhalf* p_enc_Whh_b = (bhalf*)alloc((size_t)256 * 1024 * 2);
  bhalf* p_fc_h      = (bhalf*)alloc((size_t)512 * 512 * 2);
  bhalf* p_fc_c      = (bhalf*)alloc((size_t)512 * 512 * 2);
  bhalf* p_top_f     = (bhalf*)alloc((size_t)512 * 512 * 2);
  bhalf* p_bot_f     = (bhalf*)alloc((size_t)512 * 512 * 2);
  bhalf* p_top_b     = (bhalf*)alloc((size_t)512 * 512 * 2);
  bhalf* p_bot_b     = (bhalf*)alloc((size_t)512 * 512 * 2);
  bhalf* p_dec_Wih_f = (bhalf*)alloc((size_t)1024 * 2048 * 2);
  bhalf* p_dec_Whh_f = (bhalf*)alloc((size_t)512 * 2048 * 2);
  bhalf* p_dec_Wih_b = (bhalf*)alloc((size_t)1024 * 2048 * 2);
  bhalf* p_dec_Whh_b = (bhalf*)alloc((size_t)512 * 2048 * 2);
  bhalf* p_fc2       = (bhalf*)alloc((size_t)1024 * 1024 * 2);
  // f32 activations
  float* ea      = (float*)alloc((size_t)128 * 8 * 512 * 4);
  float* es      = (float*)alloc((size_t)2 * 32 * 8 * 512 * 4);
  float* enc_out = (float*)alloc((size_t)128 * 8 * 512 * 4);
  float* hfin    = (float*)alloc((size_t)2 * 8 * 256 * 4);
  float* h0      = (float*)alloc((size_t)8 * 512 * 4);
  float* c0      = (float*)alloc((size_t)8 * 512 * 4);
  float* EP      = (float*)alloc((size_t)2 * 1024 * 512 * 4);
  float* outs    = (float*)alloc((size_t)2 * 32 * 8 * 512 * 4);
  float* hid     = (float*)alloc((size_t)256 * 1024 * 4);

  float* out = (float*)d_out;            // [tags 256][scores 8][ob 8][feats 2048]
  float* o_tags   = out;
  float* o_scores = out + 256;
  float* o_bin    = out + 264;
  float* o_feats  = out + 272;

  // 1) pack weights to WMMA bf16 tiles
  k_pack<<<(512 * 1024 + 255) / 256, 256, 0, stream>>>(enc_Wih_f, p_enc_Wih_f, 512, 1024);
  k_pack<<<(256 * 1024 + 255) / 256, 256, 0, stream>>>(enc_Whh_f, p_enc_Whh_f, 256, 1024);
  k_pack<<<(512 * 1024 + 255) / 256, 256, 0, stream>>>(enc_Wih_b, p_enc_Wih_b, 512, 1024);
  k_pack<<<(256 * 1024 + 255) / 256, 256, 0, stream>>>(enc_Whh_b, p_enc_Whh_b, 256, 1024);
  k_pack<<<(512 * 512 + 255) / 256, 256, 0, stream>>>(fc_h_W, p_fc_h, 512, 512);
  k_pack<<<(512 * 512 + 255) / 256, 256, 0, stream>>>(fc_c_W, p_fc_c, 512, 512);
  k_pack<<<(512 * 512 + 255) / 256, 256, 0, stream>>>(attn_W_f, p_top_f, 512, 512);
  k_pack<<<(512 * 512 + 255) / 256, 256, 0, stream>>>(attn_W_f + (size_t)512 * 512, p_bot_f, 512, 512);
  k_pack<<<(512 * 512 + 255) / 256, 256, 0, stream>>>(attn_W_b, p_top_b, 512, 512);
  k_pack<<<(512 * 512 + 255) / 256, 256, 0, stream>>>(attn_W_b + (size_t)512 * 512, p_bot_b, 512, 512);
  k_pack<<<(1024 * 2048 + 255) / 256, 256, 0, stream>>>(dec_Wih_f, p_dec_Wih_f, 1024, 2048);
  k_pack<<<(512 * 2048 + 255) / 256, 256, 0, stream>>>(dec_Whh_f, p_dec_Whh_f, 512, 2048);
  k_pack<<<(1024 * 2048 + 255) / 256, 256, 0, stream>>>(dec_Wih_b, p_dec_Wih_b, 1024, 2048);
  k_pack<<<(512 * 2048 + 255) / 256, 256, 0, stream>>>(dec_Whh_b, p_dec_Whh_b, 512, 2048);
  k_pack<<<(1024 * 1024 + 255) / 256, 256, 0, stream>>>(fc2_W, p_fc2, 1024, 1024);

  // 2) embedding gathers
  k_embed<<<(128 * 8 * 512 + 2 * 32 * 8 * 512 + 255) / 256, 256, 0, stream>>>(article, summary, emb, ea, es);

  // 3) BiLSTM encoder (both directions in parallel blocks)
  k_encoder<<<2, 256, 0, stream>>>(ea, p_enc_Wih_f, p_enc_Whh_f, enc_b_f,
                                   p_enc_Wih_b, p_enc_Whh_b, enc_b_b, enc_out, hfin);

  // 4) h0/c0
  k_h0c0<<<1, 256, 0, stream>>>(hfin, p_fc_h, fc_h_b, p_fc_c, fc_c_b, h0, c0);

  // 5) hoisted attention precompute (enc_out @ W_bot, both dirs)
  k_attnpre<<<128, 256, 0, stream>>>(enc_out, p_bot_f, p_bot_b, EP);

  // 6) attention decoders
  k_decoder<<<2, 256, 0, stream>>>(es, h0, c0, EP, enc_out,
                                   p_top_f, p_top_b, attn_v_f, attn_v_b,
                                   p_dec_Wih_f, p_dec_Wih_b, p_dec_Whh_f, p_dec_Whh_b,
                                   dec_b_f, dec_b_b, outs);

  // 7) head
  k_fc2<<<16, 256, 0, stream>>>(outs, p_fc2, fc2_b, hid);
  k_feats<<<8, 256, 0, stream>>>(outs, qlabel, tag_W, tag_b, o_feats);
  k_bin<<<1, 256, 0, stream>>>(hid, bin_W, bin_b, o_bin);

  // 8) constrained Viterbi
  k_viterbi<<<1, 32, 0, stream>>>(o_feats, summary, crf_trans, crf_start, crf_end,
                                  o_tags, o_scores);
}
